// EfficientAttentionScore_41412074668294
// MI455X (gfx1250) — compile-verified, matrix operand order checked by reading, not run on hardware
//
#include <hip/hip_runtime.h>
#include <hip/hip_bf16.h>
#include <math.h>

// ---------------------------------------------------------------------------
// EfficientAttentionScore for MI455X (gfx1250, wave32, WMMA bf16 path)
//   keys   = softmax(Wk x + bk, axis=s)
//   values = Wv x + bv
//   out    = keys . values^T   (per batch)
//
// Pipeline:
//   K1: keys_raw / values GEMM (bf16 WMMA, fp32 accum) -> bf16 in d_ws
//   K2: per-row softmax, normalized exp written back in place (bf16)
//   K3: context GEMM (pure bf16 WMMA, fp32 accum) -> fp32 d_out
//
// All LDS tiles use row stride 40 bf16 (80 B): 16B-aligned rows for
// ds_load_b128 fragments and conflict-free bank mapping (20 dwords/row).
// ---------------------------------------------------------------------------

typedef __attribute__((ext_vector_type(16))) __bf16 v16bf;
typedef __attribute__((ext_vector_type(8)))  __bf16 v8bf;
typedef __attribute__((ext_vector_type(4)))  __bf16 v4bf;
typedef __attribute__((ext_vector_type(8)))  float  v8f;
typedef __attribute__((ext_vector_type(4)))  float  v4f;
typedef __bf16 bf16_t;

#define B_   16
#define D_   384
#define HW_  4096
#define KC_  384
#define VC_  384
#define LDP  40   // padded LDS row stride (bf16 elems): 80B, 16B-aligned, bank-clean

// ---- WMMA 16x16x32 bf16 fragment gather ----
// Tile stored [row][k] with stride LDP. Lane = 16*half + m.
// Lane's 16 elems: k in {0..7}+8*half and {16..23}+8*half -> two b128 loads.
__device__ __forceinline__ v16bf frag16(const bf16_t* lds, int row0, int lane) {
    int m = lane & 15, half = lane >> 4;
    const bf16_t* p = lds + (row0 + m) * LDP + half * 8;
    v8bf lo = *(const v8bf*)p;
    v8bf hi = *(const v8bf*)(p + 16);
    return __builtin_shufflevector(lo, hi, 0, 1, 2, 3, 4, 5, 6, 7,
                                           8, 9, 10, 11, 12, 13, 14, 15);
}

#define WMMA_BF16(A, Bm, C) \
    __builtin_amdgcn_wmma_f32_16x16x32_bf16(false, (A), false, (Bm), (short)0, (C), false, false)

// ---------------------------------------------------------------------------
// Kernel 1: keys_raw / values GEMM.  grid = (HW/64, B, 2); block = 256 (8 waves)
// Workgroup: full-M (384) x 64-col tile of (W x[b] + bias), K chunked by 32.
// Wave w: rows [48w, 48w+48): 3 M-tiles x 4 N-tiles.
// ---------------------------------------------------------------------------
__global__ void __launch_bounds__(256)
kv_gemm_kernel(const float* __restrict__ x,
               const float* __restrict__ Wk, const float* __restrict__ bk,
               const float* __restrict__ Wv, const float* __restrict__ bv,
               bf16_t* __restrict__ keys, bf16_t* __restrict__ values)
{
    __shared__ bf16_t sA[KC_ * LDP];  // W chunk   [384 rows][32 k] (padded)
    __shared__ bf16_t sB[64 * LDP];   // x chunk^T [64 n][32 k]     (padded)

    const int tid  = threadIdx.x;
    const int lane = tid & 31;
    const int wave = tid >> 5;
    const int n0   = blockIdx.x * 64;
    const int b    = blockIdx.y;
    const int z    = blockIdx.z;

    const float* W    = z ? Wv : Wk;
    const float* bias = z ? bv : bk;
    bf16_t*      out  = z ? values : keys;

    v8f acc[3][4] = {};

    for (int kk = 0; kk < D_; kk += 32) {
        // stage W[., kk:kk+32] as bf16, float4-vectorized (16B global loads)
        for (int q = tid; q < KC_ * 8; q += 256) {
            int r = q >> 3, c = (q & 7) << 2;
            v4f w = *(const v4f*)&W[r * D_ + kk + c];
            v4bf o;
            o[0] = (bf16_t)w[0]; o[1] = (bf16_t)w[1];
            o[2] = (bf16_t)w[2]; o[3] = (bf16_t)w[3];
            *(v4bf*)&sA[r * LDP + c] = o;
        }
        // stage x[b, kk:kk+32, n0:n0+64] transposed -> sB[n][k]
        // (coalesced global reads along n; padded stride keeps stores bank-clean)
        for (int i = tid; i < 32 * 64; i += 256) {
            int c = i & 63, r = i >> 6;
            const float* gp = &x[((size_t)b * D_ + kk + r) * HW_ + n0 + c];
            sB[c * LDP + r] = (bf16_t)(*gp);
            if (kk + 32 < D_) __builtin_prefetch(gp + 32 * HW_, 0, 0);
        }
        __syncthreads();

        v16bf bf[4];
#pragma unroll
        for (int nt = 0; nt < 4; ++nt) bf[nt] = frag16(sB, nt * 16, lane);
#pragma unroll
        for (int mt = 0; mt < 3; ++mt) {
            v16bf af = frag16(sA, wave * 48 + mt * 16, lane);
#pragma unroll
            for (int nt = 0; nt < 4; ++nt)
                acc[mt][nt] = WMMA_BF16(af, bf[nt], acc[mt][nt]);
        }
        __syncthreads();
    }

    // epilogue: + bias, store bf16.  C/D layout: row = r + 8*half, col = lane&15
    const int half = lane >> 4, n = lane & 15;
#pragma unroll
    for (int mt = 0; mt < 3; ++mt) {
#pragma unroll
        for (int nt = 0; nt < 4; ++nt) {
#pragma unroll
            for (int r = 0; r < 8; ++r) {
                int row = wave * 48 + mt * 16 + r + half * 8;
                int col = n0 + nt * 16 + n;
                float v = acc[mt][nt][r] + bias[row];
                out[((size_t)b * KC_ + row) * HW_ + col] = (bf16_t)v;
            }
        }
    }
}

// ---------------------------------------------------------------------------
// Kernel 2: softmax over each 4096-long row, normalized exp written in place.
// grid = B*KC blocks; each block owns one row exclusively (in-place is safe).
// ---------------------------------------------------------------------------
__global__ void __launch_bounds__(256)
softmax_kernel(bf16_t* __restrict__ keys)
{
    __shared__ float red[256];
    const int tid = threadIdx.x;
    bf16_t* p = keys + (size_t)blockIdx.x * HW_;

    float m = -INFINITY;
    for (int i = tid; i < HW_; i += 256) m = fmaxf(m, (float)p[i]);
    red[tid] = m; __syncthreads();
    for (int s = 128; s > 0; s >>= 1) {
        if (tid < s) red[tid] = fmaxf(red[tid], red[tid + s]);
        __syncthreads();
    }
    m = red[0]; __syncthreads();

    float sum = 0.0f;
    for (int i = tid; i < HW_; i += 256) sum += __expf((float)p[i] - m);
    red[tid] = sum; __syncthreads();
    for (int s = 128; s > 0; s >>= 1) {
        if (tid < s) red[tid] += red[tid + s];
        __syncthreads();
    }
    const float rcp = 1.0f / red[0];

    for (int i = tid; i < HW_; i += 256)
        p[i] = (bf16_t)(__expf((float)p[i] - m) * rcp);
}

// ---------------------------------------------------------------------------
// Kernel 3: context = keys . values^T  (both bf16, softmax already applied).
// grid = (3, 3, B): 128x128 output tiles, K = HW chunked by 32 through LDS.
// Per-batch operands (6.3 MB bf16) are L2-resident across tile re-reads.
// 8 waves in 4x2 grid; wave tile = 32 rows x 64 cols (2x4 WMMA tiles).
// ---------------------------------------------------------------------------
__global__ void __launch_bounds__(256)
context_gemm_kernel(const bf16_t* __restrict__ keys,
                    const bf16_t* __restrict__ values,
                    float* __restrict__ out)
{
    __shared__ bf16_t sA[128 * LDP];  // keys tile   [128 m][32 s] (padded)
    __shared__ bf16_t sB[128 * LDP];  // values tile [128 v][32 s] (padded)

    const int tid  = threadIdx.x;
    const int lane = tid & 31;
    const int wave = tid >> 5;
    const int wm   = wave & 3;        // 4 wave-rows (32 rows each)
    const int wn   = wave >> 2;       // 2 wave-cols (64 cols each)
    const int b    = blockIdx.z;
    const int m0   = blockIdx.x * 128;
    const int n0   = blockIdx.y * 128;

    v8f acc[2][4] = {};

    for (int kk = 0; kk < HW_; kk += 32) {
        // both tiles: contiguous along s in global and LDS -> 16B copies
        for (int v = tid; v < 512; v += 256) {
            int r = v >> 2, k8 = (v & 3) << 3;
            *(v8bf*)&sA[r * LDP + k8] =
                *(const v8bf*)&keys[((size_t)b * KC_ + m0 + r) * HW_ + kk + k8];
        }
        for (int v = tid; v < 512; v += 256) {
            int r = v >> 2, k8 = (v & 3) << 3;
            *(v8bf*)&sB[r * LDP + k8] =
                *(const v8bf*)&values[((size_t)b * VC_ + n0 + r) * HW_ + kk + k8];
        }
        __syncthreads();

        v16bf bf[4];
#pragma unroll
        for (int nt = 0; nt < 4; ++nt) bf[nt] = frag16(sB, wn * 64 + nt * 16, lane);
#pragma unroll
        for (int mt = 0; mt < 2; ++mt) {
            v16bf af = frag16(sA, wm * 32 + mt * 16, lane);
#pragma unroll
            for (int nt = 0; nt < 4; ++nt)
                acc[mt][nt] = WMMA_BF16(af, bf[nt], acc[mt][nt]);
        }
        __syncthreads();
    }

    const int half = lane >> 4, n = lane & 15;
#pragma unroll
    for (int mt = 0; mt < 2; ++mt) {
#pragma unroll
        for (int nt = 0; nt < 4; ++nt) {
#pragma unroll
            for (int r = 0; r < 8; ++r) {
                int row = m0 + wm * 32 + mt * 16 + r + half * 8;
                int col = n0 + wn * 64 + nt * 16 + n;
                out[((size_t)b * KC_ + row) * VC_ + col] = acc[mt][nt][r];
            }
        }
    }
}

// ---------------------------------------------------------------------------
extern "C" void kernel_launch(void* const* d_in, const int* in_sizes, int n_in,
                              void* d_out, int out_size, void* d_ws, size_t ws_size,
                              hipStream_t stream)
{
    const float* x  = (const float*)d_in[0];
    const float* Wk = (const float*)d_in[1];
    const float* bk = (const float*)d_in[2];
    const float* Wv = (const float*)d_in[3];
    const float* bv = (const float*)d_in[4];
    // d_in[5], d_in[6] are h, w scalars (compile-time constants here)
    float* out = (float*)d_out;

    // workspace: keys bf16 (raw, then normalized in place) | values bf16
    const size_t kvElems = (size_t)B_ * KC_ * HW_;
    bf16_t* keys   = (bf16_t*)d_ws;
    bf16_t* values = keys + kvElems;

    kv_gemm_kernel<<<dim3(HW_ / 64, B_, 2), 256, 0, stream>>>(
        x, Wk, bk, Wv, bv, keys, values);
    softmax_kernel<<<dim3(B_ * KC_), 256, 0, stream>>>(keys);
    context_gemm_kernel<<<dim3(KC_ / 128, VC_ / 128, B_), 256, 0, stream>>>(
        keys, values, out);
}
